// FPNBasedRPN_2576980377791
// MI455X (gfx1250) — matrix operand order repaired and checked by reading, not compiled, not run on hardware
//
#include <hip/hip_runtime.h>
#include <hip/hip_bf16.h>
#include <math.h>

typedef __attribute__((ext_vector_type(16))) __bf16 v16bf;
typedef __attribute__((ext_vector_type(8)))  float  v8f;
typedef __attribute__((ext_vector_type(4)))  unsigned uint4v;
typedef __attribute__((ext_vector_type(8)))  unsigned uint8v;

#define NEGV (-1000000000.0f)
#define NTOT 261888   // 3 * (256^2 + 128^2 + 64^2 + 32^2 + 16^2)

// padded channel-last bf16 feature buffer: per level (H+2)*(W+2)*256 elems
#define FB_L2 0
#define FB_L3 17040384   // 258*258*256
#define FB_L4 21366784   // + 130*130*256
#define FB_L5 22481920   // + 66*66*256
#define FB_L6 22777856   // + 34*34*256
#define FB_TOT 22860800  // + 18*18*256

__device__ __forceinline__ unsigned short f32_to_bf16_rne(float f) {
    unsigned int u = __float_as_uint(f);
    unsigned int r = u + 0x7FFFu + ((u >> 16) & 1u);
    return (unsigned short)(r >> 16);
}

// ---------------------------------------------------------------------------
// Kernel 1a: repack conv_w (OIHW 256x256x3x3 f32) -> bf16 [tap][co][ci]
// ---------------------------------------------------------------------------
__global__ void k_pack_weights(const float* __restrict__ w,
                               unsigned short* __restrict__ wbf) {
    int d = blockIdx.x * blockDim.x + threadIdx.x;   // d = (t*256+co)*256+ci
    if (d >= 256 * 256 * 9) return;
    int ci = d & 255;
    int co = (d >> 8) & 255;
    int t  = d >> 16;
    wbf[d] = f32_to_bf16_rne(w[(co * 256 + ci) * 9 + t]);
}

// ---------------------------------------------------------------------------
// Kernel 1b: features NCHW f32 -> bf16 channel-last, 1px zero-padded border,
// layout per level: [H+2][W+2][256]. This makes each conv halo tile a clean
// 3-D TDM descriptor tile (dim0=ci contiguous, dim1=cols, dim2=rows).
// ---------------------------------------------------------------------------
__global__ void k_pack_feats(const float* __restrict__ p2, const float* __restrict__ p3,
                             const float* __restrict__ p4, const float* __restrict__ p5,
                             const float* __restrict__ p6,
                             unsigned short* __restrict__ fb) {
    int d = blockIdx.x * blockDim.x + threadIdx.x;
    if (d >= FB_TOT) return;
    const float* fin; int H; int l;
    if (d < FB_L3)      { fin = p2; H = 256; l = d - FB_L2; }
    else if (d < FB_L4) { fin = p3; H = 128; l = d - FB_L3; }
    else if (d < FB_L5) { fin = p4; H = 64;  l = d - FB_L4; }
    else if (d < FB_L6) { fin = p5; H = 32;  l = d - FB_L5; }
    else                { fin = p6; H = 16;  l = d - FB_L6; }
    int W = H, Wp = W + 2;
    int ci  = l & 255;
    int pix = l >> 8;
    int pc  = pix % Wp;
    int pr  = pix / Wp;
    float v = 0.0f;
    if (pr >= 1 && pr <= H && pc >= 1 && pc <= W)
        v = fin[(size_t)ci * H * W + (size_t)(pr - 1) * W + (pc - 1)];
    fb[d] = f32_to_bf16_rne(v);
}

// ---------------------------------------------------------------------------
// TDM issue: 3-D tile (32 ci x 34 cols x 3 rows), bf16, into LDS at lds_off.
// D# packed per CDNA5 ISA 8.3-8.6. Linear fill order -> LDS [row][col][ci].
// ---------------------------------------------------------------------------
__device__ __forceinline__ void tdm_load_halo(const unsigned short* gaddr,
                                              unsigned lds_off,
                                              int rem_cols, int rem_rows,
                                              int row_stride_elems) {
    unsigned long long ga = (unsigned long long)(uintptr_t)gaddr;
    uint4v g0;
    g0.x = 1u;                                   // count=1, no gather/iterate
    g0.y = lds_off;                              // lds_addr (bytes)
    g0.z = (unsigned)(ga & 0xFFFFFFFFull);       // global_addr[31:0]
    g0.w = (unsigned)((ga >> 32) & 0x01FFFFFFull) | 0x80000000u; // addr[56:32] | type=2

    uint8v g1;
    g1.s0 = (1u << 16);                          // workgroup_mask=0, data_size=1 (2B)
    g1.s1 = (256u & 0xFFFFu) << 16;              // tensor_dim0 = 256 (lo16)
    g1.s2 = ((unsigned)rem_cols & 0xFFFFu) << 16;// tensor_dim0 hi=0 | tensor_dim1 lo16
    g1.s3 = (32u << 16);                         // tensor_dim1 hi=0 | tile_dim0=32
    g1.s4 = 34u | (3u << 16);                    // tile_dim1=34 | tile_dim2=3
    g1.s5 = 256u;                                // tensor_dim0_stride lo32 (ci->col)
    unsigned rs = (unsigned)row_stride_elems;    // tensor_dim1_stride (col->row)
    g1.s6 = (rs & 0xFFFFu) << 16;                // d0_stride hi16=0 | d1_stride lo16
    g1.s7 = rs >> 16;                            // d1_stride [47:16]

    uint4v g2;
    g2.x = (unsigned)rem_rows;                   // tensor_dim2
    g2.y = 0u;                                   // tensor_dim3 (unused)
    g2.z = 0u;                                   // tensor_dim2_stride lo32
    g2.w = 0u;                                   // d2_stride hi | tile_dim3=0
    uint4v g3; g3.x = 0u; g3.y = 0u; g3.z = 0u; g3.w = 0u;

    asm volatile("tensor_load_to_lds %0, %1, %2, %3"
                 :: "s"(g0), "s"(g1), "s"(g2), "s"(g3)
                 : "memory");
}

// ---------------------------------------------------------------------------
// Kernel 2: fused 3x3 conv (implicit GEMM, bf16 WMMA, TDM-staged B tiles)
//           + ReLU + 1x1 loc/score + softmax fg + anchor decode + filter.
// One WG = 256 threads (8 waves) per (level, y, 32-wide x tile).
// ---------------------------------------------------------------------------
__global__ void __launch_bounds__(256)
k_rpn_conv(const unsigned short* __restrict__ fb,
           const unsigned short* __restrict__ wbf,
           const float* __restrict__ conv_b,
           const float* __restrict__ loc_w, const float* __restrict__ loc_b,
           const float* __restrict__ score_w, const float* __restrict__ score_b,
           float* __restrict__ scores_ws, float* __restrict__ boxes_ws)
{
    __shared__ unsigned short sBd[2][3 * 34 * 32]; // double-buffered halo tiles
    __shared__ float sH[256 * 32];                 // hidden activations [M][n]
    __shared__ float sLoc[32 * 12];
    __shared__ float sSc[32 * 6];

    // ---- decode block -> (level, y, x0) ----
    int b = blockIdx.x;
    int H, stride, aoff; size_t fboff;
    if (b < 2048)      { H = 256; stride = 4;  aoff = 0;      fboff = FB_L2; }
    else if (b < 2560) { H = 128; stride = 8;  aoff = 196608; fboff = FB_L3; b -= 2048; }
    else if (b < 2688) { H = 64;  stride = 16; aoff = 245760; fboff = FB_L4; b -= 2560; }
    else if (b < 2720) { H = 32;  stride = 32; aoff = 258048; fboff = FB_L5; b -= 2688; }
    else               { H = 16;  stride = 64; aoff = 261120; fboff = FB_L6; b -= 2720; }
    int W   = H, Wp = W + 2;
    int nxt = (W + 31) >> 5;
    int y   = b / nxt;
    int x0  = (b % nxt) << 5;

    int tid    = threadIdx.x;
    int lane   = tid & 31;
    int wv     = tid >> 5;          // wave id 0..7 -> M rows [wv*32, wv*32+31]
    int laneHi = (lane >> 4) & 1;   // lane half (ISA WMMA layout split)
    int lane15 = lane & 15;

    const unsigned short* lev = fb + fboff;
    unsigned lds0 = (unsigned)(uintptr_t)(void*)&sBd[0][0]; // flat low32 == LDS offset
    unsigned lds1 = (unsigned)(uintptr_t)(void*)&sBd[1][0];

    v8f zero8 = {0.f, 0.f, 0.f, 0.f, 0.f, 0.f, 0.f, 0.f};
    v8f acc[2][2];
    acc[0][0] = zero8; acc[0][1] = zero8; acc[1][0] = zero8; acc[1][1] = zero8;

    int rem_cols = Wp - x0;          // TDM OOB-zeros cols past tensor edge (p6)
    int rem_rows = Wp - y;           // always >= 3 (padded)

    // prologue: TDM chunk 0 -> buffer 0
    if (wv == 0) {
        const unsigned short* g =
            lev + ((size_t)(y * Wp + x0) * 256 + 0);
        tdm_load_halo(g, lds0, rem_cols, rem_rows, Wp * 256);
    }

    // ---- K loop: 8 chunks of 32 input channels, double-buffered TDM ----
    for (int i = 0; i < 8; ++i) {
        __syncthreads();   // readers of buf[(i+1)&1] (iteration i-1) are done
        if (wv == 0) {
            if (i < 7) {
                const unsigned short* g =
                    lev + ((size_t)(y * Wp + x0) * 256 + (i + 1) * 32);
                tdm_load_halo(g, (i & 1) ? lds0 : lds1, rem_cols, rem_rows, Wp * 256);
                __builtin_amdgcn_s_wait_tensorcnt(1);  // chunk i landed
            } else {
                __builtin_amdgcn_s_wait_tensorcnt(0);
            }
        }
        __syncthreads();   // publish chunk i's LDS tile to all waves
        const unsigned short* sB = sBd[i & 1];
        int kc = i * 32;

        for (int t = 0; t < 9; ++t) {       // 3x3 taps
            int dy = t / 3;                 // LDS row 0..2  (y offset -1..+1)
            int dx = t % 3;                 // LDS col base  (x offset -1..+1)

            // B fragments (K=32 x N=16), contiguous k = 16*laneHi + s
            union { unsigned u[8]; v16bf v; } Bt[2];
#pragma unroll
            for (int ni = 0; ni < 2; ++ni) {
                int n = ni * 16 + lane15;
                const unsigned* basep =
                    (const unsigned*)&sB[(dy * 34 + dx + n) * 32 + laneHi * 16];
#pragma unroll
                for (int vv = 0; vv < 8; ++vv) Bt[ni].u[vv] = basep[vv];
            }

            // A fragments (M=16 x K=32), contiguous k = 16*laneHi + s
            union { unsigned u[8]; v16bf v; } At[2];
#pragma unroll
            for (int mi = 0; mi < 2; ++mi) {
                int M = wv * 32 + mi * 16 + lane15;
                const unsigned short* arow =
                    wbf + ((size_t)(t * 256 + M)) * 256 + kc;
#pragma unroll
                for (int vv = 0; vv < 8; ++vv) {
                    int k0 = (vv * 2) + (laneHi << 4);
                    At[mi].u[vv] = *(const unsigned*)(arow + k0);
                }
            }

#pragma unroll
            for (int mi = 0; mi < 2; ++mi)
#pragma unroll
                for (int ni = 0; ni < 2; ++ni)
                    acc[mi][ni] = __builtin_amdgcn_wmma_f32_16x16x32_bf16(
                        false, At[mi].v, false, Bt[ni].v,
                        (short)0, acc[mi][ni], false, false);
        }
    }

    // ---- epilogue: bias + ReLU -> sH (C/D layout: M = r + (lane>=16)*8) ----
    __syncthreads();
#pragma unroll
    for (int mi = 0; mi < 2; ++mi)
#pragma unroll
        for (int ni = 0; ni < 2; ++ni)
#pragma unroll
            for (int r = 0; r < 8; ++r) {
                int M = wv * 32 + mi * 16 + laneHi * 8 + r;
                int n = ni * 16 + lane15;
                float v = acc[mi][ni][r] + conv_b[M];
                sH[M * 32 + n] = v > 0.0f ? v : 0.0f;
            }
    __syncthreads();

    // ---- 1x1 heads: 32 pixels x (12 loc + 6 score) dot products ----
    for (int idx = tid; idx < 32 * 18; idx += 256) {
        int n = idx / 18;
        int j = idx % 18;
        const float* wp;
        float s;
        if (j < 12) { wp = loc_w + j * 256;          s = loc_b[j]; }
        else        { wp = score_w + (j - 12) * 256; s = score_b[j - 12]; }
        for (int c = 0; c < 256; ++c) s += sH[c * 32 + n] * wp[c];
        if (j < 12) sLoc[n * 12 + j] = s;
        else        sSc[n * 6 + (j - 12)] = s;
    }
    __syncthreads();

    // ---- softmax fg + anchor decode + clip + min-size ----
    for (int idx = tid; idx < 96; idx += 256) {
        int n = idx / 3, a = idx % 3;
        int x = x0 + n;
        if (x >= W) continue;
        float ratio = (a == 0) ? 0.5f : (a == 1 ? 1.0f : 2.0f);
        float sr  = sqrtf(ratio);
        float hs  = (float)stride * 8.0f * sr;
        float wsz = (float)stride * 8.0f / sr;
        float cy0 = (float)y * (float)stride;
        float cx0 = (float)x * (float)stride;
        float dyv = sLoc[n * 12 + a * 4 + 0];
        float dxv = sLoc[n * 12 + a * 4 + 1];
        float dh  = sLoc[n * 12 + a * 4 + 2];
        float dw  = sLoc[n * 12 + a * 4 + 3];
        float cy  = dyv * hs + cy0;
        float cx  = dxv * wsz + cx0;
        float hh  = expf(dh) * hs;
        float ww  = expf(dw) * wsz;
        float b0  = fminf(fmaxf(cy - 0.5f * hh, 0.f), 1024.f);
        float b1  = fminf(fmaxf(cx - 0.5f * ww, 0.f), 1024.f);
        float b2  = fminf(fmaxf(cy + 0.5f * hh, 0.f), 1024.f);
        float b3  = fminf(fmaxf(cx + 0.5f * ww, 0.f), 1024.f);
        float s0  = sSc[n * 6 + a * 2 + 0];
        float s1  = sSc[n * 6 + a * 2 + 1];
        float fg  = 1.0f / (1.0f + expf(s0 - s1));
        if (!((b2 - b0) >= 16.0f && (b3 - b1) >= 16.0f)) fg = NEGV;
        int g = aoff + ((y * W + x) * 3 + a);
        scores_ws[g]       = fg;
        boxes_ws[g * 4 + 0] = b0;
        boxes_ws[g * 4 + 1] = b1;
        boxes_ws[g * 4 + 2] = b2;
        boxes_ws[g * 4 + 3] = b3;
    }
}

// ---------------------------------------------------------------------------
// Kernel 3: iterative argmax NMS, 300 rounds, one 1024-thread workgroup.
// ---------------------------------------------------------------------------
__global__ void __launch_bounds__(1024)
k_nms(const float* __restrict__ boxes, float* __restrict__ scores,
      float* __restrict__ out)
{
    __shared__ float s_bv[1024];
    __shared__ int   s_bi[1024];
    __shared__ float s_box[4];
    __shared__ int   s_info[2];   // [0]=sel idx, [1]=valid

    int tid = threadIdx.x;
    for (int it = 0; it < 300; ++it) {
        float bv = -3.0e38f; int bi = NTOT;
        for (int j = tid; j < NTOT; j += 1024) {
            float v = scores[j];
            if (v > bv) { bv = v; bi = j; }
        }
        s_bv[tid] = bv; s_bi[tid] = bi;
        __syncthreads();
        for (int off = 512; off > 0; off >>= 1) {
            if (tid < off) {
                float ov = s_bv[tid + off]; int oi = s_bi[tid + off];
                if (ov > s_bv[tid] || (ov == s_bv[tid] && oi < s_bi[tid])) {
                    s_bv[tid] = ov; s_bi[tid] = oi;
                }
            }
            __syncthreads();
        }
        if (tid == 0) {
            int i = s_bi[0];
            int valid = s_bv[0] > (NEGV * 0.5f);
            s_info[0] = i; s_info[1] = valid;
            float b0 = boxes[i * 4 + 0], b1 = boxes[i * 4 + 1];
            float b2 = boxes[i * 4 + 2], b3 = boxes[i * 4 + 3];
            s_box[0] = b0; s_box[1] = b1; s_box[2] = b2; s_box[3] = b3;
            float kf = valid ? 1.0f : 0.0f;
            out[it * 4 + 0] = b0 * kf; out[it * 4 + 1] = b1 * kf;
            out[it * 4 + 2] = b2 * kf; out[it * 4 + 3] = b3 * kf;
        }
        __syncthreads();
        if (s_info[1]) {
            float A0 = s_box[0], A1 = s_box[1], A2 = s_box[2], A3 = s_box[3];
            float areaA = (A2 - A0) * (A3 - A1);
            int sel = s_info[0];
            for (int j = tid; j < NTOT; j += 1024) {
                float B0 = boxes[j * 4 + 0], B1 = boxes[j * 4 + 1];
                float B2 = boxes[j * 4 + 2], B3 = boxes[j * 4 + 3];
                float ty = fmaxf(A0, B0), tx = fmaxf(A1, B1);
                float by = fminf(A2, B2), bx = fminf(A3, B3);
                float inter = fmaxf(by - ty, 0.f) * fmaxf(bx - tx, 0.f);
                float areaB = (B2 - B0) * (B3 - B1);
                float iou = inter / fmaxf(areaA + areaB - inter, 1e-9f);
                if (iou > 0.7f || j == sel) scores[j] = NEGV;
            }
        }
        __syncthreads();
    }
}

// ---------------------------------------------------------------------------
extern "C" void kernel_launch(void* const* d_in, const int* in_sizes, int n_in,
                              void* d_out, int out_size, void* d_ws, size_t ws_size,
                              hipStream_t stream) {
    const float* p2      = (const float*)d_in[0];
    const float* p3      = (const float*)d_in[1];
    const float* p4      = (const float*)d_in[2];
    const float* p5      = (const float*)d_in[3];
    const float* p6      = (const float*)d_in[4];
    const float* conv_w  = (const float*)d_in[5];
    const float* conv_b  = (const float*)d_in[6];
    const float* loc_w   = (const float*)d_in[7];
    const float* loc_b   = (const float*)d_in[8];
    const float* score_w = (const float*)d_in[9];
    const float* score_b = (const float*)d_in[10];

    // workspace layout (~52 MB):
    //   scores (NTOT f32) | boxes (4*NTOT f32) | wbf (256*256*9 bf16)
    //   | featbf: padded channel-last bf16 pyramid (FB_TOT bf16)
    float* ws = (float*)d_ws;
    float* scores = ws;
    float* boxes  = ws + NTOT;
    unsigned short* wbf    = (unsigned short*)(ws + (size_t)NTOT * 5);
    unsigned short* featbf = wbf + 256 * 256 * 9;

    k_pack_weights<<<(256 * 256 * 9 + 255) / 256, 256, 0, stream>>>(conv_w, wbf);
    k_pack_feats<<<FB_TOT / 256, 256, 0, stream>>>(p2, p3, p4, p5, p6, featbf);
    k_rpn_conv<<<2736, 256, 0, stream>>>(featbf, wbf, conv_b,
                                         loc_w, loc_b, score_w, score_b,
                                         scores, boxes);
    k_nms<<<1, 1024, 0, stream>>>(boxes, scores, (float*)d_out);
}
